// DualSignalLinkPredictorC_64493228917220
// MI455X (gfx1250) — matrix operand order repaired and checked by reading, not compiled, not run on hardware
//
#include <hip/hip_runtime.h>
#include <hip/hip_bf16.h>

// ---- CDNA5 WMMA vector types (probe-confirmed signatures) ----
typedef __attribute__((ext_vector_type(16))) __bf16 v16bf;
typedef __attribute__((ext_vector_type(8)))  __bf16 v8bf;
typedef __attribute__((ext_vector_type(8)))  float  v8f;

#define LDSP 40   // padded bf16 elems per LDS row (80B => stride 20 banks, conflict-free)

// float -> bf16 bits, round-to-nearest-even
__device__ __forceinline__ unsigned short f2bf(float f) {
  unsigned u = __float_as_uint(f);
  unsigned r = u + 0x7FFFu + ((u >> 16) & 1u);
  return (unsigned short)(r >> 16);
}
// monotonic f32 <-> ordered-u32 (for atomicMax on floats)
__device__ __forceinline__ unsigned f2ord(float f) {
  unsigned u = __float_as_uint(f);
  return (u & 0x80000000u) ? ~u : (u | 0x80000000u);
}
__device__ __forceinline__ float ord2f(unsigned u) {
  return (u & 0x80000000u) ? __uint_as_float(u & 0x7FFFFFFFu) : __uint_as_float(~u);
}
__device__ __forceinline__ float nfix(float v) {   // nan/inf -> 0
  return (v * 0.0f == 0.0f) ? v : 0.0f;
}

// ---------------- weight convert f32 -> bf16 ----------------
__global__ void cvt_bf16_kernel(const float* __restrict__ in,
                                unsigned short* __restrict__ out, int n) {
  int i = blockIdx.x * 256 + threadIdx.x;
  if (i < n) out[i] = f2bf(in[i]);
}

__global__ void fill_u32_kernel(unsigned* __restrict__ p, unsigned v, int n) {
  int i = blockIdx.x * 256 + threadIdx.x;
  if (i < n) p[i] = v;
}

// ---------------- WMMA GEMM: out[N,C] = A[N,K] @ W[C,K]^T (+bias) ----------------
// block = 256 threads (8 waves). Tile: 32 rows x (64*CT) cols per block.
// wave w: rowtile (w>>2), colgroup (w&3) -> CT accumulators (CT WMMAs per A fragment).
// A tile (32x32 bf16) double-buffered in LDS, software-pipelined:
//   bf16 path: GLOBAL_LOAD_ASYNC_TO_LDS_B64 for tile k+1 issued while WMMAs run
//              on tile k; s_wait_asynccnt deferred to next iteration's top.
//   f32 path:  manual stage with fused nan_to_num + bf16 convert (same pipeline).
template <typename TA, int CT>
__global__ __launch_bounds__(256) void gemm_wmma_kernel(
    const TA* __restrict__ A, const unsigned short* __restrict__ W,
    const float* __restrict__ bias, float* __restrict__ out,
    int K, int C, int nanfix) {
  __shared__ unsigned short As[2][32 * LDSP];
  const int tid = threadIdx.x, lane = tid & 31, wave = tid >> 5;
  const int rowbase = blockIdx.x * 32;
  const int r0 = (wave >> 2) * 16;
  const int c0 = blockIdx.y * (64 * CT) + (wave & 3) * (16 * CT);
  const int arow = tid >> 3;        // 0..31
  const int akq  = (tid & 7) * 4;   // 0..28
  const int m = lane & 15, hi = lane >> 4;
  // LDS byte addresses of this thread's staging slots (ISA 10.2: generic-ptr[31:0] == LDS addr)
  const unsigned lds_slot0 = (unsigned)(uintptr_t)(&As[0][arow * LDSP + akq]);
  const unsigned lds_slot1 = (unsigned)(uintptr_t)(&As[1][arow * LDSP + akq]);

  v8f acc[CT];
  {
    v8f z = {};
#pragma unroll
    for (int i = 0; i < CT; i++) acc[i] = z;
  }

  auto stage = [&](int kb, int buf) {
    if (sizeof(TA) == 4) {
      const float* Af = (const float*)A;
      const float* src = Af + (size_t)(rowbase + arow) * K + kb + akq;
      float x0 = src[0], x1 = src[1], x2 = src[2], x3 = src[3];
      if (nanfix) { x0 = nfix(x0); x1 = nfix(x1); x2 = nfix(x2); x3 = nfix(x3); }
      unsigned short* dd = &As[buf][arow * LDSP + akq];
      dd[0] = f2bf(x0); dd[1] = f2bf(x1); dd[2] = f2bf(x2); dd[3] = f2bf(x3);
    } else {
      const unsigned short* Ab = (const unsigned short*)A;
      const unsigned short* g = Ab + (size_t)(rowbase + arow) * K + kb + akq;
      // async DMA: global -> LDS, 8 bytes per lane, tracked on ASYNCcnt (no wait here)
      asm volatile("global_load_async_to_lds_b64 %0, %1, off"
                   :: "v"(buf ? lds_slot1 : lds_slot0),
                      "v"((unsigned long long)(uintptr_t)g)
                   : "memory");
    }
  };

  stage(0, 0);                       // prologue: fill buffer 0
  int cur = 0;
  for (int kb = 0; kb < K; kb += 32, cur ^= 1) {
    if (sizeof(TA) != 4)
      asm volatile("s_wait_asynccnt 0x0" ::: "memory");   // tile kb landed (this wave)
    __syncthreads();                                       // ... and all waves' slices
    if (kb + 32 < K) stage(kb + 32, cur ^ 1);              // overlap DMA with compute

    // ---- A fragment (16x32 bf16): lane<16 -> K 0..7 & 16..23, lane>=16 -> 8..15 & 24..31
    const __bf16* arp = (const __bf16*)&As[cur][(r0 + m) * LDSP + hi * 8];
    v8bf a0 = *(const v8bf*)arp;
    v8bf a1 = *(const v8bf*)(arp + 16);
    v16bf af;
#pragma unroll
    for (int i = 0; i < 8; i++) { af[i] = a0[i]; af[i + 8] = a1[i]; }

    // ---- CT B fragments, CT WMMAs reusing the single A fragment
#pragma unroll
    for (int ct = 0; ct < CT; ct++) {
      const __bf16* wp0 = (const __bf16*)(W + (size_t)(c0 + ct * 16 + m) * K + kb + hi * 8);
      if (kb + 32 < K)
        __builtin_prefetch((const void*)(wp0 + 32), 0, 1);   // global_prefetch_b8
      v8bf b0 = *(const v8bf*)wp0;
      v8bf b1 = *(const v8bf*)(wp0 + 16);
      v16bf bfr;
#pragma unroll
      for (int i = 0; i < 8; i++) { bfr[i] = b0[i]; bfr[i + 8] = b1[i]; }
      acc[ct] = __builtin_amdgcn_wmma_f32_16x16x32_bf16(false, af, false, bfr,
                                                        (short)0, acc[ct], false, false);
    }
    __syncthreads();                 // readers done before buffer is re-staged
  }

#pragma unroll
  for (int ct = 0; ct < CT; ct++) {
    const int col = c0 + ct * 16 + m;
    const float bv = bias ? bias[col] : 0.0f;
#pragma unroll
    for (int r = 0; r < 8; r++) {
      const int row = rowbase + r0 + hi * 8 + r;   // C/D layout: lane<16 -> M=r, lane>=16 -> M=8+r
      out[(size_t)row * C + col] = acc[ct][r] + bv;
    }
  }
}

// ---------------- LayerNorm + ReLU -> bf16 (D=256, wave per row) ----------------
__global__ __launch_bounds__(256) void ln_relu_kernel(
    const float* __restrict__ in, const float* __restrict__ prebias,
    const float* __restrict__ g, const float* __restrict__ b,
    unsigned short* __restrict__ out, int n) {
  const int lane = threadIdx.x & 31, wave = threadIdx.x >> 5;
  const int node = blockIdx.x * 8 + wave;
  if (node >= n) return;
  const float* row = in + (size_t)node * 256 + lane * 8;
  float v[8];
  {
    float4 p0 = *(const float4*)(row);
    float4 p1 = *(const float4*)(row + 4);
    v[0] = p0.x; v[1] = p0.y; v[2] = p0.z; v[3] = p0.w;
    v[4] = p1.x; v[5] = p1.y; v[6] = p1.z; v[7] = p1.w;
  }
  if (prebias) {
#pragma unroll
    for (int i = 0; i < 8; i++) v[i] += prebias[lane * 8 + i];
  }
  float s = 0.f, ss = 0.f;
#pragma unroll
  for (int i = 0; i < 8; i++) { s += v[i]; ss += v[i] * v[i]; }
  for (int off = 16; off; off >>= 1) {
    s  += __shfl_xor(s, off, 32);
    ss += __shfl_xor(ss, off, 32);
  }
  const float mu  = s * (1.0f / 256.0f);
  const float var = ss * (1.0f / 256.0f) - mu * mu;
  const float inv = rsqrtf(var + 1e-5f);
#pragma unroll
  for (int i = 0; i < 8; i++) {
    const int c = lane * 8 + i;
    float y = (v[i] - mu) * inv * g[c] + b[c];
    out[(size_t)node * 256 + c] = f2bf(fmaxf(y, 0.0f));
  }
}

// ---------------- add optional bias, store f32 vec, compute clamped L2 norm (D=128) ----
__global__ __launch_bounds__(256) void finalize_kernel(
    const float* __restrict__ in, const float* __restrict__ bias,
    float* __restrict__ outv, float* __restrict__ outn, int n) {
  const int lane = threadIdx.x & 31, wave = threadIdx.x >> 5;
  const int node = blockIdx.x * 8 + wave;
  if (node >= n) return;
  float4 v = *(const float4*)(in + (size_t)node * 128 + lane * 4);
  if (bias) {
    const float4 bb = *(const float4*)(bias + lane * 4);
    v.x += bb.x; v.y += bb.y; v.z += bb.z; v.w += bb.w;
  }
  *(float4*)(outv + (size_t)node * 128 + lane * 4) = v;
  float ss = v.x * v.x + v.y * v.y + v.z * v.z + v.w * v.w;
  for (int off = 16; off; off >>= 1) ss += __shfl_xor(ss, off, 32);
  if (lane == 0) outn[node] = fmaxf(sqrtf(ss), 1e-8f);
}

// ---------------- GAT edge pass A: scores + segment_max (wave per edge) ----------------
template <int D, int H>
__global__ __launch_bounds__(256) void gat_score_kernel(
    const float* __restrict__ xl, const float* __restrict__ xr,
    const int* __restrict__ ei, const float* __restrict__ att,
    float* __restrict__ score, unsigned* __restrict__ maxb, int E, int Nn) {
  const int EPL = D / 32;
  const int lane = threadIdx.x & 31, wave = threadIdx.x >> 5;
  const int e = blockIdx.x * 8 + wave;
  const int NE = E + Nn;
  if (e >= NE) return;
  int s, d;
  if (e < E) { s = ei[e]; d = ei[E + e]; } else { s = d = e - E; }
  const float* pl = xl + (size_t)s * D + lane * EPL;
  const float* pr = xr + (size_t)d * D + lane * EPL;
  float sc = 0.f;
#pragma unroll
  for (int c = 0; c < EPL; c += 4) {
    const float4 a = *(const float4*)(pl + c);
    const float4 b = *(const float4*)(pr + c);
    const float4 w = *(const float4*)(att + lane * EPL + c);
    float u0 = a.x + b.x, u1 = a.y + b.y, u2 = a.z + b.z, u3 = a.w + b.w;
    u0 = (u0 > 0.f) ? u0 : 0.2f * u0;      // leaky_relu(0.2)
    u1 = (u1 > 0.f) ? u1 : 0.2f * u1;
    u2 = (u2 > 0.f) ? u2 : 0.2f * u2;
    u3 = (u3 > 0.f) ? u3 : 0.2f * u3;
    sc += w.x * u0 + w.y * u1 + w.z * u2 + w.w * u3;
  }
  const int G = 32 / H;                     // lanes per head
  for (int off = G >> 1; off; off >>= 1) sc += __shfl_xor(sc, off, 32);
  if ((lane & (G - 1)) == 0) {
    const int h = lane / G;
    score[(size_t)e * H + h] = sc;
    atomicMax(&maxb[(size_t)d * H + h], f2ord(sc));
  }
}

// ---------------- GAT edge pass B: exp + segment_sum denom ----------------
__global__ __launch_bounds__(256) void gat_exp_kernel(
    float* __restrict__ score, const unsigned* __restrict__ maxb,
    float* __restrict__ denom, const int* __restrict__ ei,
    int E, int Nn, int H) {
  const int idx = blockIdx.x * 256 + threadIdx.x;
  const int tot = (E + Nn) * H;
  if (idx >= tot) return;
  const int e = idx / H, h = idx - e * H;
  const int d = (e < E) ? ei[E + e] : (e - E);
  const float mx = ord2f(maxb[(size_t)d * H + h]);
  const float ex = __expf(score[idx] - mx);
  score[idx] = ex;                                   // overwrite score with exp
  unsafeAtomicAdd(&denom[(size_t)d * H + h], ex);    // GLOBAL_ATOMIC_ADD_F32
}

// ---------------- GAT edge pass C: weighted aggregation (wave per edge) ----------------
template <int D, int H>
__global__ __launch_bounds__(256) void gat_agg_kernel(
    const float* __restrict__ xl, const float* __restrict__ ex,
    const float* __restrict__ denom, const int* __restrict__ ei,
    float* __restrict__ agg, int E, int Nn) {
  const int EPL = D / 32;
  const int lane = threadIdx.x & 31, wave = threadIdx.x >> 5;
  const int e = blockIdx.x * 8 + wave;
  const int NE = E + Nn;
  if (e >= NE) return;
  int s, d;
  if (e < E) { s = ei[e]; d = ei[E + e]; } else { s = d = e - E; }
  const int h = (lane * EPL) / (D / H);
  const float a = ex[(size_t)e * H + h] / (denom[(size_t)d * H + h] + 1e-16f);
  const float* pl = xl + (size_t)s * D + lane * EPL;
  float* pa = agg + (size_t)d * D + lane * EPL;
#pragma unroll
  for (int c = 0; c < EPL; c += 4) {
    const float4 v = *(const float4*)(pl + c);
    unsafeAtomicAdd(&pa[c + 0], a * v.x);
    unsafeAtomicAdd(&pa[c + 1], a * v.y);
    unsafeAtomicAdd(&pa[c + 2], a * v.z);
    unsafeAtomicAdd(&pa[c + 3], a * v.w);
  }
}

// ---------------- final: cosine pair scores (wave per pair, D=128) ----------------
__global__ __launch_bounds__(256) void pair_kernel(
    const float* __restrict__ zg, const float* __restrict__ zf,
    const float* __restrict__ ng, const float* __restrict__ nf,
    const int* __restrict__ pairs, const float* __restrict__ la,
    const float* __restrict__ temp, float* __restrict__ out, int P) {
  const int lane = threadIdx.x & 31, wave = threadIdx.x >> 5;
  const int p = blockIdx.x * 8 + wave;
  if (p >= P) return;
  const int ps = pairs[2 * p], pd = pairs[2 * p + 1];
  const float4 ga = *(const float4*)(zg + (size_t)ps * 128 + lane * 4);
  const float4 gb = *(const float4*)(zg + (size_t)pd * 128 + lane * 4);
  const float4 fa = *(const float4*)(zf + (size_t)ps * 128 + lane * 4);
  const float4 fb = *(const float4*)(zf + (size_t)pd * 128 + lane * 4);
  float dg = ga.x * gb.x + ga.y * gb.y + ga.z * gb.z + ga.w * gb.w;
  float df = fa.x * fb.x + fa.y * fb.y + fa.z * fb.z + fa.w * fb.w;
  for (int off = 16; off; off >>= 1) {
    dg += __shfl_xor(dg, off, 32);
    df += __shfl_xor(df, off, 32);
  }
  if (lane == 0) {
    const float gs = dg / (ng[ps] * ng[pd]);
    const float fs = df / (nf[ps] * nf[pd]);
    const float alpha = 1.0f / (1.0f + __expf(-la[0]));
    out[p] = (alpha * gs + (1.0f - alpha) * fs) * temp[0];
  }
}

// ============================ host launcher ============================
extern "C" void kernel_launch(void* const* d_in, const int* in_sizes, int n_in,
                              void* d_out, int out_size, void* d_ws, size_t ws_size,
                              hipStream_t stream) {
  const float* x     = (const float*)d_in[0];
  const int*   ei    = (const int*)d_in[1];
  const int*   pairs = (const int*)d_in[2];
  const float* Wp    = (const float*)d_in[3];
  const float* bp    = (const float*)d_in[4];
  const float* g0    = (const float*)d_in[5];
  const float* b0    = (const float*)d_in[6];
  const float* Wl1   = (const float*)d_in[7];
  const float* Wr1   = (const float*)d_in[8];
  const float* att1  = (const float*)d_in[9];
  const float* bias1 = (const float*)d_in[10];
  const float* g1    = (const float*)d_in[11];
  const float* be1   = (const float*)d_in[12];
  const float* Wl2   = (const float*)d_in[13];
  const float* Wr2   = (const float*)d_in[14];
  const float* att2  = (const float*)d_in[15];
  const float* bias2 = (const float*)d_in[16];
  const float* Wm1   = (const float*)d_in[17];
  const float* bm1   = (const float*)d_in[18];
  const float* gm    = (const float*)d_in[19];
  const float* bm    = (const float*)d_in[20];
  const float* Wm2   = (const float*)d_in[21];
  const float* bm2   = (const float*)d_in[22];
  const float* la    = (const float*)d_in[23];
  const float* temp  = (const float*)d_in[24];
  float* out = (float*)d_out;

  const int RAW = 512, IN = 256, HID = 256, EMB = 128;
  const int Nn = in_sizes[0] / RAW;
  const int E  = in_sizes[1] / 2;
  const int P  = in_sizes[2] / 2;
  const int NE = E + Nn;

  char* ws = (char*)d_ws;
  size_t off = 0;
  auto alloc = [&](size_t bytes) -> char* {
    size_t r = off;
    off += (bytes + 255) & ~(size_t)255;
    return ws + r;
  };

  unsigned short* wp16  = (unsigned short*)alloc((size_t)IN  * RAW * 2);
  unsigned short* wl1b  = (unsigned short*)alloc((size_t)HID * IN  * 2);
  unsigned short* wr1b  = (unsigned short*)alloc((size_t)HID * IN  * 2);
  unsigned short* wm1b  = (unsigned short*)alloc((size_t)HID * IN  * 2);
  unsigned short* wl2b  = (unsigned short*)alloc((size_t)EMB * HID * 2);
  unsigned short* wr2b  = (unsigned short*)alloc((size_t)EMB * HID * 2);
  unsigned short* wm2b  = (unsigned short*)alloc((size_t)EMB * HID * 2);
  float*          tmpf  = (float*)alloc((size_t)Nn * HID * 4);   // pre-LN / agg1
  unsigned short* xprojb= (unsigned short*)alloc((size_t)Nn * IN * 2);
  float*          bufA  = (float*)alloc((size_t)Nn * HID * 4);   // hidpre -> xl1 -> xl2
  float*          bufB  = (float*)alloc((size_t)Nn * HID * 4);   // xr1 -> xr2
  unsigned short* hb    = (unsigned short*)alloc((size_t)Nn * HID * 2);
  unsigned short* hidb  = (unsigned short*)alloc((size_t)Nn * HID * 2);
  float*          zfeat = (float*)alloc((size_t)Nn * EMB * 4);
  float*          zg    = (float*)alloc((size_t)Nn * EMB * 4);
  float*          score = (float*)alloc((size_t)NE * 4 * 4);
  unsigned*       maxb  = (unsigned*)alloc((size_t)Nn * 4 * 4);
  float*          denom = (float*)alloc((size_t)Nn * 4 * 4);
  float*          ngn   = (float*)alloc((size_t)Nn * 4);
  float*          nft   = (float*)alloc((size_t)Nn * 4);
  (void)ws_size; (void)n_in; (void)out_size;

  const dim3 blk(256);
  auto cvt = [&](const float* s, unsigned short* d, int n) {
    cvt_bf16_kernel<<<(n + 255) / 256, blk, 0, stream>>>(s, d, n);
  };
  cvt(Wp, wp16, IN * RAW);
  cvt(Wl1, wl1b, HID * IN);  cvt(Wr1, wr1b, HID * IN);  cvt(Wm1, wm1b, HID * IN);
  cvt(Wl2, wl2b, EMB * HID); cvt(Wr2, wr2b, EMB * HID); cvt(Wm2, wm2b, EMB * HID);

  // x_proj = relu(LN(x @ Wp.T + bp))   (nan_to_num fused into A load)
  gemm_wmma_kernel<float, 4><<<dim3(Nn / 32, IN / 256), blk, 0, stream>>>(
      x, wp16, bp, tmpf, RAW, IN, 1);
  ln_relu_kernel<<<(Nn + 7) / 8, blk, 0, stream>>>(tmpf, nullptr, g0, b0, xprojb, Nn);

  // MLP branch: z_feat = relu(LN(x_proj@Wm1.T+bm1)) @ Wm2.T + bm2
  gemm_wmma_kernel<unsigned short, 4><<<dim3(Nn / 32, HID / 256), blk, 0, stream>>>(
      xprojb, wm1b, bm1, bufA, IN, HID, 0);
  ln_relu_kernel<<<(Nn + 7) / 8, blk, 0, stream>>>(bufA, nullptr, gm, bm, hidb, Nn);
  gemm_wmma_kernel<unsigned short, 2><<<dim3(Nn / 32, EMB / 128), blk, 0, stream>>>(
      hidb, wm2b, bm2, zfeat, HID, EMB, 0);
  finalize_kernel<<<(Nn + 7) / 8, blk, 0, stream>>>(zfeat, nullptr, zfeat, nft, Nn);

  // GAT layer 1 (4 heads x 64)
  gemm_wmma_kernel<unsigned short, 4><<<dim3(Nn / 32, HID / 256), blk, 0, stream>>>(
      xprojb, wl1b, nullptr, bufA, IN, HID, 0);
  gemm_wmma_kernel<unsigned short, 4><<<dim3(Nn / 32, HID / 256), blk, 0, stream>>>(
      xprojb, wr1b, nullptr, bufB, IN, HID, 0);
  fill_u32_kernel<<<(Nn * 4 + 255) / 256, blk, 0, stream>>>(maxb, 0x007FFFFFu, Nn * 4);
  hipMemsetAsync(denom, 0, (size_t)Nn * 4 * 4, stream);
  hipMemsetAsync(tmpf, 0, (size_t)Nn * HID * 4, stream);
  gat_score_kernel<256, 4><<<(NE + 7) / 8, blk, 0, stream>>>(
      bufA, bufB, ei, att1, score, maxb, E, Nn);
  gat_exp_kernel<<<(NE * 4 + 255) / 256, blk, 0, stream>>>(
      score, maxb, denom, ei, E, Nn, 4);
  gat_agg_kernel<256, 4><<<(NE + 7) / 8, blk, 0, stream>>>(
      bufA, score, denom, ei, tmpf, E, Nn);
  ln_relu_kernel<<<(Nn + 7) / 8, blk, 0, stream>>>(tmpf, bias1, g1, be1, hb, Nn);

  // GAT layer 2 (1 head x 128)
  gemm_wmma_kernel<unsigned short, 2><<<dim3(Nn / 32, EMB / 128), blk, 0, stream>>>(
      hb, wl2b, nullptr, bufA, HID, EMB, 0);
  gemm_wmma_kernel<unsigned short, 2><<<dim3(Nn / 32, EMB / 128), blk, 0, stream>>>(
      hb, wr2b, nullptr, bufB, HID, EMB, 0);
  fill_u32_kernel<<<(Nn + 255) / 256, blk, 0, stream>>>(maxb, 0x007FFFFFu, Nn);
  hipMemsetAsync(denom, 0, (size_t)Nn * 4, stream);
  hipMemsetAsync(zg, 0, (size_t)Nn * EMB * 4, stream);
  gat_score_kernel<128, 1><<<(NE + 7) / 8, blk, 0, stream>>>(
      bufA, bufB, ei, att2, score, maxb, E, Nn);
  gat_exp_kernel<<<(NE + 255) / 256, blk, 0, stream>>>(
      score, maxb, denom, ei, E, Nn, 1);
  gat_agg_kernel<128, 1><<<(NE + 7) / 8, blk, 0, stream>>>(
      bufA, score, denom, ei, zg, E, Nn);
  finalize_kernel<<<(Nn + 7) / 8, blk, 0, stream>>>(zg, bias2, zg, ngn, Nn);

  // pair scores
  pair_kernel<<<(P + 7) / 8, blk, 0, stream>>>(
      zg, zfeat, ngn, nft, pairs, la, temp, out, P);
}
